// M2RAGNN_63694365000396
// MI455X (gfx1250) — matrix operand order repaired and checked by reading, not compiled, not run on hardware
//
#include <hip/hip_runtime.h>
#include <hip/hip_bf16.h>
#include <math.h>

typedef _Float16 v16h __attribute__((ext_vector_type(16)));
typedef _Float16 h2v  __attribute__((ext_vector_type(2)));
typedef float    v8f  __attribute__((ext_vector_type(8)));

#define GN 100000
#define GE 300000
#define GH 128

// ---------------------------------------------------------------------------
// GEMM: C[N,128] = A[N,128] @ W[128,128] (+bias) (+=C if accumulate)
// One wave computes a 16x128 output strip via 8 accumulator tiles,
// K=128 consumed as 4 x v_wmma_f32_16x16x32_f16 per tile.
// W is staged transposed into LDS as f16 (stride 130 to avoid bank conflicts).
// ---------------------------------------------------------------------------
__global__ void __launch_bounds__(256)
gemm128_wmma(const float* __restrict__ A, const float* __restrict__ W,
             const float* __restrict__ bias, float* __restrict__ C,
             int N, int accumulate)
{
    __shared__ _Float16 WT[128 * 130];
    for (int i = threadIdx.x; i < 128 * 128; i += 256) {
        int k = i >> 7, n = i & 127;
        WT[n * 130 + k] = (_Float16)W[i];
    }
    __syncthreads();

    int lane = threadIdx.x & 31;
    int wave = blockIdx.x * (blockDim.x >> 5) + (threadIdx.x >> 5);
    int row0 = wave * 16;
    if (row0 >= N) return;

    int m  = lane & 15;   // row within tile (A) / col within tile (B,C)
    int hi = lane >> 4;   // half-wave select

    v8f acc[8];
    #pragma unroll
    for (int t = 0; t < 8; ++t)
        acc[t] = (v8f){0.f, 0.f, 0.f, 0.f, 0.f, 0.f, 0.f, 0.f};

    const float* arow = A + (size_t)(row0 + m) * GH;

    #pragma unroll
    for (int kk = 0; kk < 4; ++kk) {
        const int kbase = kk * 32;
        // A operand: 16x32 f16, ISA 7.12.2 layout
        v16h a;
        #pragma unroll
        for (int j = 0; j < 8; ++j) {
            int kb = kbase + ((j >> 2) << 4) + (hi << 3) + ((j & 3) << 1);
            float2 av = *(const float2*)(arow + kb);
            a[2 * j]     = (_Float16)av.x;
            a[2 * j + 1] = (_Float16)av.y;
        }
        #pragma unroll
        for (int nt = 0; nt < 8; ++nt) {
            const _Float16* wrow = &WT[(nt * 16 + m) * 130];
            v16h b;
            #pragma unroll
            for (int j = 0; j < 8; ++j) {
                h2v bv = *(const h2v*)(wrow + kbase + ((j >> 2) << 4) + (hi << 3) + ((j & 3) << 1));
                b[2 * j]     = bv[0];
                b[2 * j + 1] = bv[1];
            }
            acc[nt] = __builtin_amdgcn_wmma_f32_16x16x32_f16(
                false, a, false, b, (short)0, acc[nt], false, false);
        }
    }

    // C/D layout: VGPR i -> row = row0 + hi*8 + i, col = nt*16 + m
    #pragma unroll
    for (int nt = 0; nt < 8; ++nt) {
        #pragma unroll
        for (int i = 0; i < 8; ++i) {
            int rr = row0 + hi * 8 + i;
            int cc = nt * 16 + m;
            float v = acc[nt][i];
            if (bias) v += bias[cc];
            size_t idx = (size_t)rr * GH + cc;
            if (accumulate) v += C[idx];
            C[idx] = v;
        }
    }
}

// ---------------------------------------------------------------------------
// Attention source/dest logits: ES[n,h] = sum_f H[n,h,f]*a_s[h,f], same for ED.
// One wave per node; per-head shfl reduction.
// ---------------------------------------------------------------------------
__global__ void alpha_kernel(const float* __restrict__ H,
                             const float* __restrict__ a_s,
                             const float* __restrict__ a_d,
                             float* __restrict__ ES, float* __restrict__ ED,
                             int N, int heads)
{
    int wave = (blockIdx.x * blockDim.x + threadIdx.x) >> 5;
    int lane = threadIdx.x & 31;
    if (wave >= N) return;
    int f0 = lane * 4;
    float4 hv = *(const float4*)(H + (size_t)wave * GH + f0);
    float4 sv = *(const float4*)(a_s + f0);
    float4 dv = *(const float4*)(a_d + f0);
    float ps = hv.x * sv.x + hv.y * sv.y + hv.z * sv.z + hv.w * sv.w;
    float pd = hv.x * dv.x + hv.y * dv.y + hv.z * dv.z + hv.w * dv.w;
    int Fh = GH / heads;
    int group = Fh / 4;  // lanes per head
    for (int off = group >> 1; off > 0; off >>= 1) {
        ps += __shfl_xor(ps, off, 32);
        pd += __shfl_xor(pd, off, 32);
    }
    if ((lane & (group - 1)) == 0) {
        int head = f0 / Fh;
        ES[(size_t)wave * heads + head] = ps;
        ED[(size_t)wave * heads + head] = pd;
    }
}

__device__ __forceinline__ void atomicMaxF(float* addr, float val)
{
    if (val >= 0.f) atomicMax((int*)addr, __float_as_int(val));
    else            atomicMin((unsigned int*)addr, __float_as_uint(val));
}

// e = leaky_relu(ES[row]+ED[col]); EBUF=e; MMAX[col] = max(...)
__global__ void edge_logits_max(const int* __restrict__ row, const int* __restrict__ col,
                                const float* __restrict__ ES, const float* __restrict__ ED,
                                float* __restrict__ EBUF, float* __restrict__ MMAX,
                                int E, int N, int heads)
{
    int idx = blockIdx.x * blockDim.x + threadIdx.x;
    int total = (E + N) * heads;
    if (idx >= total) return;
    int e = idx / heads, h = idx - e * heads;
    int r, c;
    if (e < E) { r = row[e]; c = col[e]; } else { r = c = e - E; }
    float x = ES[(size_t)r * heads + h] + ED[(size_t)c * heads + h];
    x = (x >= 0.f) ? x : 0.2f * x;
    EBUF[idx] = x;
    atomicMaxF(&MMAX[(size_t)c * heads + h], x);
}

// ex = exp(e - MMAX[col]); EBUF=ex; Z[col] += ex
__global__ void edge_exp_sum(const int* __restrict__ col,
                             float* __restrict__ EBUF, const float* __restrict__ MMAX,
                             float* __restrict__ Z, int E, int N, int heads)
{
    int idx = blockIdx.x * blockDim.x + threadIdx.x;
    int total = (E + N) * heads;
    if (idx >= total) return;
    int e = idx / heads, h = idx - e * heads;
    int c = (e < E) ? col[e] : (e - E);
    float ex = expf(EBUF[idx] - MMAX[(size_t)c * heads + h]);
    EBUF[idx] = ex;
    atomicAdd(&Z[(size_t)c * heads + h], ex);
}

// OUT[col,:] += H[row,:] * (ex / (Z[col]+1e-16)); one wave per edge.
__global__ void edge_message(const int* __restrict__ row, const int* __restrict__ col,
                             const float* __restrict__ HF, const float* __restrict__ EX,
                             const float* __restrict__ Z, float* __restrict__ OUT,
                             int E, int N, int heads)
{
    int wave = (blockIdx.x * blockDim.x + threadIdx.x) >> 5;
    int lane = threadIdx.x & 31;
    if (wave >= E + N) return;
    int r, c;
    if (wave < E) { r = row[wave]; c = col[wave]; } else { r = c = wave - E; }
    int f0 = lane * 4;
    int Fh = GH / heads;
    int head = f0 / Fh;
    float a = EX[(size_t)wave * heads + head] /
              (Z[(size_t)c * heads + head] + 1e-16f);
    float4 hv = *(const float4*)(HF + (size_t)r * GH + f0);
    float* o = OUT + (size_t)c * GH + f0;
    atomicAdd(o + 0, hv.x * a);
    atomicAdd(o + 1, hv.y * a);
    atomicAdd(o + 2, hv.z * a);
    atomicAdd(o + 3, hv.w * a);
}

// sum of edge distances over E mol edges -> gsum[0]
__global__ void geo_reduce(const int* __restrict__ row, const int* __restrict__ col,
                           const float* __restrict__ pos, float* __restrict__ gsum, int E)
{
    int idx = blockIdx.x * blockDim.x + threadIdx.x;
    float d = 0.f;
    if (idx < E) {
        int r = row[idx], c = col[idx];
        float dx = pos[r * 3 + 0] - pos[c * 3 + 0];
        float dy = pos[r * 3 + 1] - pos[c * 3 + 1];
        float dz = pos[r * 3 + 2] - pos[c * 3 + 2];
        d = sqrtf(dx * dx + dy * dy + dz * dz);
    }
    for (int off = 16; off > 0; off >>= 1) d += __shfl_xor(d, off, 32);
    __shared__ float ws[8];
    int lane = threadIdx.x & 31, w = threadIdx.x >> 5;
    if (lane == 0) ws[w] = d;
    __syncthreads();
    if (threadIdx.x == 0) {
        float s = 0.f;
        for (int i = 0; i < 8; ++i) s += ws[i];
        atomicAdd(gsum, s);
    }
}

__global__ void fill_f32(float* __restrict__ p, float v, int count)
{
    int i = blockIdx.x * blockDim.x + threadIdx.x;
    if (i < count) p[i] = v;
}

// XM = hin + gsum[0]*scale (scale = 0.1/E)
__global__ void add_geo(const float* __restrict__ hin, const float* __restrict__ gsum,
                        float scale, float* __restrict__ XM, int count)
{
    int i = blockIdx.x * blockDim.x + threadIdx.x;
    if (i < count) XM[i] = hin[i] + gsum[0] * scale;
}

__global__ void add_bias_rows(float* __restrict__ X, const float* __restrict__ bias, int count)
{
    int i = blockIdx.x * blockDim.x + threadIdx.x;
    if (i < count) X[i] += bias[i & (GH - 1)];
}

// hout = relu(SUM + b1[col] + b2[col])
__global__ void finalize_relu(const float* __restrict__ SUM, const float* __restrict__ b1,
                              const float* __restrict__ b2, float* __restrict__ OUT, int count)
{
    int i = blockIdx.x * blockDim.x + threadIdx.x;
    if (i >= count) return;
    float v = SUM[i] + b1[i & (GH - 1)] + b2[i & (GH - 1)];
    OUT[i] = v > 0.f ? v : 0.f;
}

// out[n,0] = h2[n]·Wy + by ; out[n,1] = h2[n]·Wa + ba ; one wave per node.
__global__ void out_heads(const float* __restrict__ H2, const float* __restrict__ Wy,
                          const float* __restrict__ by, const float* __restrict__ Wa,
                          const float* __restrict__ ba, float* __restrict__ out, int N)
{
    int wave = (blockIdx.x * blockDim.x + threadIdx.x) >> 5;
    int lane = threadIdx.x & 31;
    if (wave >= N) return;
    int f0 = lane * 4;
    float4 hv = *(const float4*)(H2 + (size_t)wave * GH + f0);
    float4 wy = *(const float4*)(Wy + f0);
    float4 wa = *(const float4*)(Wa + f0);
    float py = hv.x * wy.x + hv.y * wy.y + hv.z * wy.z + hv.w * wy.w;
    float pa = hv.x * wa.x + hv.y * wa.y + hv.z * wa.z + hv.w * wa.w;
    for (int off = 16; off > 0; off >>= 1) {
        py += __shfl_xor(py, off, 32);
        pa += __shfl_xor(pa, off, 32);
    }
    if (lane == 0) {
        out[2 * (size_t)wave + 0] = py + by[0];
        out[2 * (size_t)wave + 1] = pa + ba[0];
    }
}

// ---------------------------------------------------------------------------
static inline int cdiv(long long a, int b) { return (int)((a + b - 1) / b); }

static void run_gat_conv(const float* X, const float* W, const float* as_, const float* ad_,
                         int heads, const int* row, const int* col,
                         float* HF, float* ES, float* ED, float* MMAX, float* Z, float* EBUF,
                         float* zeroFirst, float* OUTACC, hipStream_t stream)
{
    const int N = GN, E = GE;
    gemm128_wmma<<<cdiv(N / 16, 8), 256, 0, stream>>>(X, W, nullptr, HF, N, 0);
    alpha_kernel<<<cdiv((long long)N * 32, 256), 256, 0, stream>>>(HF, as_, ad_, ES, ED, N, heads);
    fill_f32<<<cdiv(N * heads, 256), 256, 0, stream>>>(MMAX, -__builtin_inff(), N * heads);
    fill_f32<<<cdiv(N * heads, 256), 256, 0, stream>>>(Z, 0.f, N * heads);
    int tot = (E + N) * heads;
    edge_logits_max<<<cdiv(tot, 256), 256, 0, stream>>>(row, col, ES, ED, EBUF, MMAX, E, N, heads);
    edge_exp_sum<<<cdiv(tot, 256), 256, 0, stream>>>(col, EBUF, MMAX, Z, E, N, heads);
    if (zeroFirst)
        fill_f32<<<cdiv(N * GH, 256), 256, 0, stream>>>(zeroFirst, 0.f, N * GH);
    edge_message<<<cdiv((long long)(E + N) * 32, 256), 256, 0, stream>>>(
        row, col, HF, EBUF, Z, OUTACC, E, N, heads);
}

extern "C" void kernel_launch(void* const* d_in, const int* in_sizes, int n_in,
                              void* d_out, int out_size, void* d_ws, size_t ws_size,
                              hipStream_t stream)
{
    (void)in_sizes; (void)n_in; (void)out_size; (void)ws_size;
    const int N = GN, E = GE;

    const float* x   = (const float*)d_in[0];
    const float* pos = (const float*)d_in[1];
    const int* ei_chem = (const int*)d_in[2];
    const int* ei_cond = (const int*)d_in[3];
    const int* ei_mol  = (const int*)d_in[4];
    const float* Wp = (const float*)d_in[5];
    const float* bp = (const float*)d_in[6];

    struct LayerP {
        const float *Wc, *asc, *adc, *bc;
        const float *Wd, *asd, *add_, *bd;
        const float *Wm, *asm_, *adm, *bm, *Wl, *bl;
    } Lp[2];
    int i = 7;
    for (int L = 0; L < 2; ++L) {
        Lp[L].Wc  = (const float*)d_in[i++]; Lp[L].asc  = (const float*)d_in[i++];
        Lp[L].adc = (const float*)d_in[i++]; Lp[L].bc   = (const float*)d_in[i++];
        Lp[L].Wd  = (const float*)d_in[i++]; Lp[L].asd  = (const float*)d_in[i++];
        Lp[L].add_= (const float*)d_in[i++]; Lp[L].bd   = (const float*)d_in[i++];
        Lp[L].Wm  = (const float*)d_in[i++]; Lp[L].asm_ = (const float*)d_in[i++];
        Lp[L].adm = (const float*)d_in[i++]; Lp[L].bm   = (const float*)d_in[i++];
        Lp[L].Wl  = (const float*)d_in[i++]; Lp[L].bl   = (const float*)d_in[i++];
    }
    const float* Wy = (const float*)d_in[i++];
    const float* by = (const float*)d_in[i++];
    const float* Wa = (const float*)d_in[i++];
    const float* ba = (const float*)d_in[i++];

    // workspace layout
    char* base = (char*)d_ws;
    size_t off = 0;
    auto take = [&](size_t bytes) -> float* {
        float* p = (float*)(base + off);
        off = (off + bytes + 255) & ~(size_t)255;
        return p;
    };
    const size_t big = (size_t)N * GH * sizeof(float);
    float* P0   = take(big);                    // h0, later h2
    float* P1   = take(big);                    // h1
    float* HF   = take(big);                    // per-conv features
    float* AGG  = take(big);                    // eqgat agg (aliases XM)
    float* SUM  = take(big);                    // layer accumulator
    float* ES   = take((size_t)N * 4 * sizeof(float));
    float* ED   = take((size_t)N * 4 * sizeof(float));
    float* MMAX = take((size_t)N * 4 * sizeof(float));
    float* Z    = take((size_t)N * 4 * sizeof(float));
    float* EBUF = take((size_t)(E + N) * 4 * sizeof(float));
    float* GS   = take(256);

    const int NH = N * GH;

    // h0 = x @ Wp + bp
    gemm128_wmma<<<cdiv(N / 16, 8), 256, 0, stream>>>(x, Wp, bp, P0, N, 0);

    // geo scalar (same for both layers: depends only on pos & mol edges)
    fill_f32<<<1, 32, 0, stream>>>(GS, 0.f, 1);
    geo_reduce<<<cdiv(E, 256), 256, 0, stream>>>(ei_mol, ei_mol + E, pos, GS, E);

    const float* hin = P0;
    float* houts[2] = { P1, P0 };
    for (int L = 0; L < 2; ++L) {
        const LayerP& p = Lp[L];
        fill_f32<<<cdiv(NH, 256), 256, 0, stream>>>(SUM, 0.f, NH);

        // chemical_transformation (heads=2) -> accumulate into SUM
        run_gat_conv(hin, p.Wc, p.asc, p.adc, 2, ei_chem, ei_chem + E,
                     HF, ES, ED, MMAX, Z, EBUF, nullptr, SUM, stream);
        // condition_modulation (heads=2) -> accumulate into SUM
        run_gat_conv(hin, p.Wd, p.asd, p.add_, 2, ei_cond, ei_cond + E,
                     HF, ES, ED, MMAX, Z, EBUF, nullptr, SUM, stream);
        // molecule_target EQGAT (heads=4): XM = hin + 0.1*mean(geo)
        add_geo<<<cdiv(NH, 256), 256, 0, stream>>>(hin, GS, 0.1f / (float)E, AGG, NH);
        run_gat_conv(AGG /*XM*/, p.Wm, p.asm_, p.adm, 4, ei_mol, ei_mol + E,
                     HF, ES, ED, MMAX, Z, EBUF, AGG /*zero after GEMM*/, AGG, stream);
        add_bias_rows<<<cdiv(NH, 256), 256, 0, stream>>>(AGG, p.bm, NH);
        // SUM += (AGG + bm) @ Wl + bl
        gemm128_wmma<<<cdiv(N / 16, 8), 256, 0, stream>>>(AGG, p.Wl, p.bl, SUM, N, 1);

        // h_next = relu(SUM + bc + bd)
        finalize_relu<<<cdiv(NH, 256), 256, 0, stream>>>(SUM, p.bc, p.bd, houts[L], NH);
        hin = houts[L];
    }

    // prediction heads: out[n,{0,1}] from h2 = P0
    out_heads<<<cdiv((long long)N * 32, 256), 256, 0, stream>>>(
        P0, Wy, by, Wa, ba, (float*)d_out, N);
}
